// RNNModel_7327214207314
// MI455X (gfx1250) — matrix operand order repaired and checked by reading, not compiled, not run on hardware
//
#include <hip/hip_runtime.h>
#include <hip/hip_bf16.h>

typedef __attribute__((ext_vector_type(16))) __bf16 v16bf;
typedef __attribute__((ext_vector_type(8)))  float  v8f;

#define BATCH 1024
#define SEQT  120
#define EMB   50
#define HID   300
#define HIDP  304          // padded hidden (19 * 16)
#define KDIM  352          // 50 + 300 + 2 pad = 11 * 32
#define NTILES 19          // HIDP / 16
#define MBLKS  16          // BATCH / 64
#define NCLS  5

__device__ __forceinline__ float sigf(float x) {
    return 1.0f / (1.0f + __expf(-x));
}
__device__ __forceinline__ float tanh_fast(float x) {
    return 2.0f * sigf(2.0f * x) - 1.0f;   // tanh(x) = 2*sigmoid(2x) - 1
}

// ---- prep: repack W_lstm (350 x 1200) -> Wt[4][HIDP][KDIM] bf16, zero padded
__global__ void prep_weights(const float* __restrict__ W, __bf16* __restrict__ Wt) {
    int idx = blockIdx.x * blockDim.x + threadIdx.x;
    if (idx >= 4 * HIDP * KDIM) return;
    int g = idx / (HIDP * KDIM);
    int r = idx % (HIDP * KDIM);
    int n = r / KDIM;
    int k = r % KDIM;
    float v = 0.0f;
    if (k < 350 && n < HID) v = W[(long)k * 1200 + g * HID + n];
    Wt[idx] = (__bf16)v;
}

// ---- prep: A0[b][k]: x_0 in cols 0..49, zeros elsewhere (h0 = 0, pad = 0)
__global__ void init_A0(const int* __restrict__ ids, const float* __restrict__ emb,
                        __bf16* __restrict__ A0) {
    int idx = blockIdx.x * blockDim.x + threadIdx.x;
    if (idx >= BATCH * KDIM) return;
    int b = idx / KDIM;
    int k = idx % KDIM;
    float v = 0.0f;
    if (k < EMB) {
        int tok = ids[(long)b * SEQT + 0];
        v = emb[(long)tok * EMB + k];
    }
    A0[idx] = (__bf16)v;
}

__global__ void init_c(float* __restrict__ C) {
    int idx = blockIdx.x * blockDim.x + threadIdx.x;
    if (idx < BATCH * HID) C[idx] = 0.0f;
}

__global__ void init_out(const float* __restrict__ b2, float* __restrict__ out) {
    int idx = blockIdx.x * blockDim.x + threadIdx.x;
    if (idx < SEQT * BATCH * NCLS) out[idx] = b2[idx % NCLS];
}

// ---- one LSTM timestep: z = A @ W (4 gates), gate math, write c, h, x_{t+1},
//      and atomically accumulate the 5-class projection into out[t].
__global__ __launch_bounds__(64) void lstm_step(
    const __bf16* __restrict__ Acur, __bf16* __restrict__ Anext,
    const __bf16* __restrict__ Wt, const float* __restrict__ bl,
    float* __restrict__ Cst, const float* __restrict__ U,
    float* __restrict__ out, const int* __restrict__ ids,
    const float* __restrict__ emb, int t)
{
    const int mblk = blockIdx.x % MBLKS;
    const int nblk = blockIdx.x / MBLKS;
    const int n0 = nblk * 16;
    const int rowbase = mblk * 64;

    // Stage x_{t+1} into Anext (one block-column does it for its 64 rows).
    if (nblk == 0 && t + 1 < SEQT) {
        int r = rowbase + (int)threadIdx.x;            // 64 threads -> 64 rows
        int tok = ids[(long)r * SEQT + (t + 1)];
        const float* e = emb + (long)tok * EMB;
        __bf16* dst = Anext + (long)r * KDIM;
        #pragma unroll
        for (int k = 0; k < EMB; ++k) dst[k] = (__bf16)e[k];
        dst[350] = (__bf16)0.0f;
        dst[351] = (__bf16)0.0f;
    }

    const int wave = threadIdx.x >> 5;
    const int lane = threadIdx.x & 31;
    const int half = lane >> 4;      // 0 or 1
    const int lidx = lane & 15;      // column within tile / row within tile
    const int r0 = rowbase + wave * 32;

    v8f acc[4][2] = {};   // [gate][mtile]

    for (int kt = 0; kt < 11; ++kt) {
        const int kb = kt * 32;
        // A fragments (16-bit A layout: half selects K groups {0-7,16-23}/{8-15,24-31})
        v16bf afrag[2];
        #pragma unroll
        for (int mt = 0; mt < 2; ++mt) {
            const __bf16* ap = Acur + (long)(r0 + mt * 16 + lidx) * KDIM + kb + half * 8;
            uint4* av = (uint4*)&afrag[mt];
            av[0] = *(const uint4*)ap;          // K = kb+half*8 .. +7
            av[1] = *(const uint4*)(ap + 16);   // K = kb+16+half*8 .. +7
        }
        #pragma unroll
        for (int g = 0; g < 4; ++g) {
            // B fragment: column n0+lidx, K = kb+half*8..+7 and kb+16+half*8..+7
            const __bf16* bp = Wt + (((long)g * HIDP + (n0 + lidx)) * KDIM) + kb + half * 8;
            v16bf bfrag;
            uint4* bv = (uint4*)&bfrag;
            bv[0] = *(const uint4*)bp;
            bv[1] = *(const uint4*)(bp + 16);
            #pragma unroll
            for (int mt = 0; mt < 2; ++mt)
                acc[g][mt] = __builtin_amdgcn_wmma_f32_16x16x32_bf16(
                    false, afrag[mt], false, bfrag, (short)0, acc[g][mt], false, false);
        }
    }

    // Epilogue: gate math per element. C/D layout: VGPR v -> row v + 8*half, col lidx.
    const int n = n0 + lidx;
    const bool nval = (n < HID);
    const float bi = nval ? bl[0 * HID + n] : 0.0f;
    const float bj = nval ? bl[1 * HID + n] : 0.0f;
    const float bf = nval ? bl[2 * HID + n] : 0.0f;
    const float bo = nval ? bl[3 * HID + n] : 0.0f;
    float Un[NCLS];
    #pragma unroll
    for (int c = 0; c < NCLS; ++c) Un[c] = nval ? U[n * NCLS + c] : 0.0f;

    #pragma unroll
    for (int mt = 0; mt < 2; ++mt) {
        const int mbase = r0 + mt * 16 + half * 8;
        #pragma unroll
        for (int v = 0; v < 8; ++v) {
            const int row = mbase + v;
            float zi = acc[0][mt][v] + bi;
            float zj = acc[1][mt][v] + bj;
            float zf = acc[2][mt][v] + bf + 1.0f;   // FORGET_BIAS
            float zo = acc[3][mt][v] + bo;
            float cold = nval ? Cst[(long)row * HID + n] : 0.0f;
            float cnew = cold * sigf(zf) + sigf(zi) * tanh_fast(zj);
            float hnew = tanh_fast(cnew) * sigf(zo);
            if (nval) {
                Cst[(long)row * HID + n] = cnew;
                if (t + 1 < SEQT)
                    Anext[(long)row * KDIM + EMB + n] = (__bf16)hnew;
            }
            // projection partial: reduce over the 16 n-lanes (halves independent)
            #pragma unroll
            for (int c = 0; c < NCLS; ++c) {
                float p = nval ? hnew * Un[c] : 0.0f;
                p += __shfl_xor(p, 1, 32);
                p += __shfl_xor(p, 2, 32);
                p += __shfl_xor(p, 4, 32);
                p += __shfl_xor(p, 8, 32);
                if (lidx == 0)
                    atomicAdd(&out[((long)t * BATCH + row) * NCLS + c], p);
            }
        }
    }
}

extern "C" void kernel_launch(void* const* d_in, const int* in_sizes, int n_in,
                              void* d_out, int out_size, void* d_ws, size_t ws_size,
                              hipStream_t stream) {
    const int*   ids = (const int*)d_in[0];
    const float* emb = (const float*)d_in[1];
    const float* W   = (const float*)d_in[2];
    const float* bl  = (const float*)d_in[3];
    const float* U   = (const float*)d_in[4];
    const float* b2  = (const float*)d_in[5];
    float* out = (float*)d_out;

    char* ws = (char*)d_ws;
    const size_t wt_bytes = (size_t)4 * HIDP * KDIM * 2;        // 856,064
    const size_t a_bytes  = (size_t)BATCH * KDIM * 2;           // 720,896
    __bf16* Wt  = (__bf16*)ws;
    __bf16* A0  = (__bf16*)(ws + wt_bytes);
    __bf16* A1  = (__bf16*)(ws + wt_bytes + a_bytes);
    float*  Cst = (float*) (ws + wt_bytes + 2 * a_bytes);       // +1,228,800 -> ~3.5 MB

    prep_weights<<<(4 * HIDP * KDIM + 255) / 256, 256, 0, stream>>>(W, Wt);
    init_A0<<<(BATCH * KDIM + 255) / 256, 256, 0, stream>>>(ids, emb, A0);
    init_c<<<(BATCH * HID + 255) / 256, 256, 0, stream>>>(Cst);
    init_out<<<(SEQT * BATCH * NCLS + 255) / 256, 256, 0, stream>>>(b2, out);

    for (int t = 0; t < SEQT; ++t) {
        const __bf16* Ac = (t & 1) ? A1 : A0;
        __bf16*       An = (t & 1) ? A0 : A1;
        lstm_step<<<MBLKS * NTILES, 64, 0, stream>>>(Ac, An, Wt, bl, Cst, U, out,
                                                     ids, emb, t);
    }
}
